// MCA_14912126452255
// MI455X (gfx1250) — matrix operand order repaired and checked by reading, not compiled, not run on hardware
//
#include <hip/hip_runtime.h>
#include <math.h>
#include <stdint.h>

typedef __attribute__((ext_vector_type(16))) __bf16 v16bf;
typedef __attribute__((ext_vector_type(8)))  __bf16 v8bf;
typedef __attribute__((ext_vector_type(8)))  float  v8f;
typedef __attribute__((ext_vector_type(4)))  unsigned int v4u;
typedef __attribute__((ext_vector_type(8)))  int v8i;
typedef __attribute__((ext_vector_type(4)))  int v4i;

// Problem constants (fixed by the reference harness)
constexpr int N  = 16;
constexpr int D  = 1024;
constexpr int T  = 1024;
constexpr int H  = 4;
constexpr int DH = 256;   // d / h
constexpr int NH = N * H; // 64

// ---------------------------------------------------------------------------
// Fragment helpers (A/B in this code's contiguous k order, not the ISA table order)
//   A 16x32 bf16: lane<16 -> row=lane, K 0..15; lane>=16 -> row=lane-16, K 16..31
//   B 32x16 bf16: lane<16 -> col=lane, K 0..15; lane>=16 -> col=lane-16, K 16..31
//   C/D f32: VGPR r <-> row r + 8*(lane>=16), col = lane&15
// ---------------------------------------------------------------------------
__device__ __forceinline__ v16bf load_a_frag(const __bf16* ap, int k0) {
  v8bf lo = *(const v8bf*)(ap + k0);
  v8bf hi = *(const v8bf*)(ap + k0 + 8);
  return __builtin_shufflevector(lo, hi, 0,1,2,3,4,5,6,7,8,9,10,11,12,13,14,15);
}
__device__ __forceinline__ v16bf load_b_frag(const __bf16* bp, int k0) {
  v8bf lo = *(const v8bf*)(bp + k0);
  v8bf hi = *(const v8bf*)(bp + k0 + 8);
  return __builtin_shufflevector(lo, hi, 0,1,2,3,4,5,6,7,8,9,10,11,12,13,14,15);
}
__device__ __forceinline__ v8f wmma_bf16(v16bf a, v16bf b, v8f c) {
  return __builtin_amdgcn_wmma_f32_16x16x32_bf16(false, a, false, b, (short)0, c,
                                                 false, false);
}

// ---------------------------------------------------------------------------
// Generic GEMM: one wave computes a 16(M) x 64(N) strip (4 accumulators,
// A fragment reused 4x).  A row-major (MxK); B stored transposed row-major
// (column n of B = contiguous K run at B + n*ldb).
// MODE: 0 = store f32, 1 = store bf16, 2 = exact GELU then store f32
// ---------------------------------------------------------------------------
template <int MODE>
__global__ void gemm_kernel(const __bf16* __restrict__ A, const __bf16* __restrict__ B,
                            void* __restrict__ Cv,
                            int Mtiles, int NtilesQ, int K,
                            int lda, int ldb, int ldc,
                            long strideA, int a_div,
                            long strideB, int b_mod,
                            long strideC, int totalQ) {
  int gwid = blockIdx.x * (blockDim.x >> 5) + (threadIdx.x >> 5);
  if (gwid >= totalQ) return;     // wave-uniform: EXEC stays all-ones
  int lane = threadIdx.x & 31;
  int nq = gwid % NtilesQ;
  int mt = (gwid / NtilesQ) % Mtiles;
  int nh = gwid / (NtilesQ * Mtiles);

  const __bf16* Ab = A + (size_t)(nh / a_div) * strideA + (size_t)(mt * 16) * lda;
  const __bf16* Bb = B + (size_t)(nh % b_mod) * strideB + (size_t)(nq * 64) * ldb;
  const __bf16* ap = Ab + (size_t)(lane & 15) * lda + ((lane >> 4) * 16);
  const __bf16* bp = Bb + (size_t)(lane & 15) * ldb + ((lane >> 4) * 16);

  v8f acc[4];
#pragma unroll
  for (int s = 0; s < 4; ++s) acc[s] = v8f{0.f,0.f,0.f,0.f,0.f,0.f,0.f,0.f};

  for (int k0 = 0; k0 < K; k0 += 32) {
    __builtin_prefetch(ap + k0 + 256, 0, 3);
    __builtin_prefetch(bp + k0 + 256, 0, 3);
    v16bf a = load_a_frag(ap, k0);
#pragma unroll
    for (int s = 0; s < 4; ++s) {
      v16bf b = load_b_frag(bp + (size_t)(s * 16) * ldb, k0);
      acc[s] = wmma_bf16(a, b, acc[s]);
    }
  }

  int col = lane & 15;
  int rbase = (lane >> 4) * 8;
  size_t cbase = (size_t)nh * strideC + (size_t)(mt * 16) * ldc + (size_t)(nq * 64) + col;
#pragma unroll
  for (int s = 0; s < 4; ++s) {
    size_t cb = cbase + s * 16;
    if constexpr (MODE == 0) {
      float* C = (float*)Cv;
#pragma unroll
      for (int r = 0; r < 8; ++r) C[cb + (size_t)(rbase + r) * ldc] = acc[s][r];
    } else if constexpr (MODE == 1) {
      __bf16* C = (__bf16*)Cv;
#pragma unroll
      for (int r = 0; r < 8; ++r) C[cb + (size_t)(rbase + r) * ldc] = (__bf16)acc[s][r];
    } else {
      float* C = (float*)Cv;
#pragma unroll
      for (int r = 0; r < 8; ++r) {
        float x = acc[s][r];
        C[cb + (size_t)(rbase + r) * ldc] = 0.5f * x * (1.0f + erff(x * 0.70710678118654752f));
      }
    }
  }
}

// ---------------------------------------------------------------------------
// TDM descriptor load: 2D tile (tile_k contiguous elems per row, tile_rows rows,
// row stride in elements), bf16 (data_size=2B), global -> LDS.
// D# layout per CDNA5 ISA ch.8 (group0 128b, group1 256b; groups 2/3 zero).
// This toolchain's builtin takes 6 args (g0, g1, g2, g3, extra v8i, cpol).
// ---------------------------------------------------------------------------
#if __has_builtin(__builtin_amdgcn_tensor_load_to_lds)
#define HAVE_TDM 1
__device__ __forceinline__ void tdm_load_2d(unsigned int lds_off, const void* gptr,
                                            unsigned int tile_k, unsigned int tile_rows,
                                            unsigned int row_stride_elems) {
  uint64_t ga = (uint64_t)(uintptr_t)gptr;
  v4u g0;
  g0[0] = 1u;                                   // count=1, user mode
  g0[1] = lds_off;                              // lds_addr (bytes)
  g0[2] = (unsigned int)ga;                     // global_addr[31:0]
  g0[3] = (unsigned int)((ga >> 32) & 0x1FFFFFFu) | (2u << 30);  // addr[56:32] | type=2
  v8i g1;
  g1[0] = (int)(1u << 16);                      // workgroup_mask=0, data_size=1 (2B)
  g1[1] = (int)((tile_k & 0xFFFFu) << 16);      // tensor_dim0[15:0] (abar addr=0)
  g1[2] = (int)(((tile_k >> 16) & 0xFFFFu) | ((tile_rows & 0xFFFFu) << 16)); // td0 hi | td1 lo
  g1[3] = (int)(((tile_rows >> 16) & 0xFFFFu) | ((tile_k & 0xFFFFu) << 16)); // td1 hi | tile_dim0
  g1[4] = (int)(tile_rows & 0xFFFFu);           // tile_dim1 (tile_dim2=0)
  g1[5] = (int)row_stride_elems;                // tensor_dim0_stride[31:0]
  g1[6] = 0;                                    // stride hi | dim1_stride lo
  g1[7] = 0;
  v4i z4 = {0, 0, 0, 0};
  v8i z8 = {0, 0, 0, 0, 0, 0, 0, 0};
  __builtin_amdgcn_tensor_load_to_lds(g0, g1, z4, z4, z8, 0);
}
#else
#define HAVE_TDM 0
#endif

__device__ __forceinline__ void wait_tensorcnt0() {
#if __has_builtin(__builtin_amdgcn_s_wait_tensorcnt)
  __builtin_amdgcn_s_wait_tensorcnt(0);
#else
  asm volatile("s_wait_tensorcnt 0x0" ::: "memory");
#endif
}

// ---------------------------------------------------------------------------
// att = W @ O_flow^T  (K = DH = 256), LDS-staged via TDM.
// Block: 256 threads (8 waves) -> 64(M) x 128(N) C-tile; waves 4(M) x 2(N),
// each wave 16x64.  A tile 64x256 bf16 (32KB) and B tile 128x256 bf16 (64KB)
// are contiguous in memory; staged whole-K into LDS once.
// ---------------------------------------------------------------------------
__global__ void att_gemm_kernel(const __bf16* __restrict__ Wm,
                                const __bf16* __restrict__ Of,
                                float* __restrict__ att) {
  extern __shared__ __align__(16) __bf16 sm[];    // [64*256] A, then [128*256] B
  __bf16* smA = sm;
  __bf16* smB = sm + 64 * DH;

  int bid = blockIdx.x;
  int nb = bid % (T / 128);
  int mb = (bid / (T / 128)) % (T / 64);
  int nh = bid / ((T / 128) * (T / 64));
  const __bf16* Abase = Wm + ((size_t)nh * T + (size_t)mb * 64) * DH;   // 64*256 contiguous
  const __bf16* Bbase = Of + ((size_t)nh * T + (size_t)nb * 128) * DH;  // 128*256 contiguous

#if HAVE_TDM
  if ((threadIdx.x >> 5) == 0) {
    unsigned int ldsA = (unsigned int)(uintptr_t)smA;
    unsigned int ldsB = (unsigned int)(uintptr_t)smB;
    tdm_load_2d(ldsA, Abase, DH, 64, DH);
    tdm_load_2d(ldsB, Bbase, DH, 128, DH);
    wait_tensorcnt0();
  }
#else
  {
    // cooperative copy fallback: both tiles are contiguous blocks
    const int chunksA = 64 * DH / 8, chunksB = 128 * DH / 8;
    for (int c = threadIdx.x; c < chunksA; c += 256)
      *(v8bf*)(smA + c * 8) = *(const v8bf*)(Abase + c * 8);
    for (int c = threadIdx.x; c < chunksB; c += 256)
      *(v8bf*)(smB + c * 8) = *(const v8bf*)(Bbase + c * 8);
  }
#endif
  __syncthreads();

  int wid = threadIdx.x >> 5;
  int lane = threadIdx.x & 31;
  int wm = wid & 3;       // 0..3  -> M offset wm*16
  int wn = wid >> 2;      // 0..1  -> N offset wn*64

  const __bf16* ap = smA + (size_t)(wm * 16 + (lane & 15)) * DH + ((lane >> 4) * 16);
  const __bf16* bp = smB + (size_t)(wn * 64 + (lane & 15)) * DH + ((lane >> 4) * 16);

  v8f acc[4];
#pragma unroll
  for (int s = 0; s < 4; ++s) acc[s] = v8f{0.f,0.f,0.f,0.f,0.f,0.f,0.f,0.f};

#pragma unroll
  for (int k0 = 0; k0 < DH; k0 += 32) {
    v16bf a = load_a_frag(ap, k0);
#pragma unroll
    for (int s = 0; s < 4; ++s) {
      v16bf b = load_b_frag(bp + (size_t)(s * 16) * DH, k0);
      acc[s] = wmma_bf16(a, b, acc[s]);
    }
  }

  int col = lane & 15;
  int rbase = (lane >> 4) * 8;
  size_t crow0 = (size_t)mb * 64 + wm * 16 + rbase;
  size_t ccol0 = (size_t)nb * 128 + wn * 64 + col;
  float* C = att + (size_t)nh * T * T;
#pragma unroll
  for (int s = 0; s < 4; ++s)
#pragma unroll
    for (int r = 0; r < 8; ++r)
      C[(crow0 + r) * T + ccol0 + s * 16] = acc[s][r];
}

// (n,d,t) f32 -> (n,t,d) bf16
__global__ void cvt_xT_kernel(const float* __restrict__ x, __bf16* __restrict__ xT) {
  int idx = blockIdx.x * blockDim.x + threadIdx.x;
  if (idx >= N * D * T) return;
  int nn = idx / (D * T);
  int rem = idx % (D * T);
  int dd = rem / T;
  int tt = rem % T;
  xT[(size_t)nn * T * D + (size_t)tt * D + dd] = (__bf16)x[idx];
}

// per-head transpose: (h, rows, cols) f32 -> (h, cols, rows) bf16
__global__ void cvt_wT_kernel(const float* __restrict__ w, __bf16* __restrict__ wT,
                              int rows, int cols, int total) {
  int idx = blockIdx.x * blockDim.x + threadIdx.x;
  if (idx >= total) return;
  int hh = idx / (rows * cols);
  int rem = idx % (rows * cols);
  int rr = rem / cols;
  int cc = rem % cols;
  wT[(size_t)hh * rows * cols + (size_t)cc * rows + rr] = (__bf16)w[idx];
}

// L2-normalize each dh-row; emit bf16 O and bf16 O^T
__global__ void norm_kernel(const float* __restrict__ O, __bf16* __restrict__ Ob,
                            __bf16* __restrict__ OTb) {
  __shared__ float smr[256];
  int row = blockIdx.x;   // 0..NH*T-1
  int e = threadIdx.x;    // DH == 256
  float v = O[(size_t)row * DH + e];
  smr[e] = v * v;
  __syncthreads();
  for (int s = 128; s > 0; s >>= 1) {
    if (e < s) smr[e] += smr[e + s];
    __syncthreads();
  }
  float inv = 1.0f / fmaxf(sqrtf(smr[0]), 1e-12f);
  float o = v * inv;
  Ob[(size_t)row * DH + e] = (__bf16)o;
  int tt = row % T;
  int nh = row / T;
  OTb[((size_t)nh * DH + e) * T + tt] = (__bf16)o;
}

__global__ void rowstats_kernel(const float* __restrict__ att,
                                float* __restrict__ rmax, float* __restrict__ rsum) {
  __shared__ float smr[256];
  int row = blockIdx.x;
  int tid = threadIdx.x;
  const float* p = att + (size_t)row * T;
  float v0 = p[tid], v1 = p[tid + 256], v2 = p[tid + 512], v3 = p[tid + 768];
  smr[tid] = fmaxf(fmaxf(v0, v1), fmaxf(v2, v3));
  __syncthreads();
  for (int s = 128; s > 0; s >>= 1) {
    if (tid < s) smr[tid] = fmaxf(smr[tid], smr[tid + s]);
    __syncthreads();
  }
  float M = smr[0];
  __syncthreads();
  smr[tid] = expf(v0 - M) + expf(v1 - M) + expf(v2 - M) + expf(v3 - M);
  __syncthreads();
  for (int s = 128; s > 0; s >>= 1) {
    if (tid < s) smr[tid] += smr[tid + s];
    __syncthreads();
  }
  if (tid == 0) { rmax[row] = M; rsum[row] = smr[0]; }
}

__global__ void colstats_kernel(const float* __restrict__ att,
                                float* __restrict__ cmax, float* __restrict__ csum) {
  __shared__ float smr[256];
  int b = blockIdx.x;       // nh*T + j
  int tid = threadIdx.x;
  int nh = b / T;
  int j = b % T;
  const float* p = att + (size_t)nh * T * T + j;
  float v0 = p[(size_t)(tid)       * T];
  float v1 = p[(size_t)(tid + 256) * T];
  float v2 = p[(size_t)(tid + 512) * T];
  float v3 = p[(size_t)(tid + 768) * T];
  smr[tid] = fmaxf(fmaxf(v0, v1), fmaxf(v2, v3));
  __syncthreads();
  for (int s = 128; s > 0; s >>= 1) {
    if (tid < s) smr[tid] = fmaxf(smr[tid], smr[tid + s]);
    __syncthreads();
  }
  float M = smr[0];
  __syncthreads();
  smr[tid] = expf(v0 - M) + expf(v1 - M) + expf(v2 - M) + expf(v3 - M);
  __syncthreads();
  for (int s = 128; s > 0; s >>= 1) {
    if (tid < s) smr[tid] += smr[tid + s];
    __syncthreads();
  }
  if (tid == 0) { cmax[b] = M; csum[b] = smr[0]; }
}

// Build P_r (row softmax, bf16) and P_f (transposed column softmax, bf16)
__global__ void pmat_kernel(const float* __restrict__ att,
                            const float* __restrict__ rmax, const float* __restrict__ rsum,
                            const float* __restrict__ cmax, const float* __restrict__ csum,
                            __bf16* __restrict__ Pr, __bf16* __restrict__ Pf) {
  long idx = (long)blockIdx.x * blockDim.x + threadIdx.x;
  if (idx >= (long)NH * T * T) return;
  long nh = idx / ((long)T * T);
  long rem = idx % ((long)T * T);
  int i = (int)(rem / T);
  int j = (int)(rem % T);
  float a = att[idx];
  long ri = nh * T + i;
  long cj = nh * T + j;
  Pr[idx] = (__bf16)(expf(a - rmax[ri]) / rsum[ri]);
  Pf[cj * T + i] = (__bf16)(expf(a - cmax[cj]) / csum[cj]);
}

// out[n,i,j] = tanh(e[n, j>>8, i, j&255] + x[n,i,j])   (e.mT.reshape identity, d==t)
__global__ void final_kernel(const float* __restrict__ x, const float* __restrict__ e,
                             float* __restrict__ out) {
  int idx = blockIdx.x * blockDim.x + threadIdx.x;
  if (idx >= N * D * T) return;
  int nn = idx / (D * T);
  int rem = idx % (D * T);
  int i = rem / T;
  int j = rem % T;
  int hh = j >> 8;
  int ee = j & 255;
  float ev = e[(((size_t)nn * H + hh) * T + i) * DH + ee];
  out[idx] = tanhf(ev + x[idx]);
}

extern "C" void kernel_launch(void* const* d_in, const int* in_sizes, int n_in,
                              void* d_out, int out_size, void* d_ws, size_t ws_size,
                              hipStream_t stream) {
  const float* rgb       = (const float*)d_in[0];
  const float* flow      = (const float*)d_in[1];
  const float* rgb_proj  = (const float*)d_in[2];
  const float* flow_proj = (const float*)d_in[3];
  const float* atte_w    = (const float*)d_in[4];
  float* out = (float*)d_out;

  char* p = (char*)d_ws;
  auto alloc = [&](size_t bytes) -> void* {
    void* r = (void*)p;
    p += (bytes + 255) & ~(size_t)255;
    return r;
  };
  const size_t NTD = (size_t)N * T * D;          // 16M elements
  __bf16* xT_rgb   = (__bf16*)alloc(NTD * 2);
  __bf16* xT_flow  = (__bf16*)alloc(NTD * 2);
  __bf16* projT_r  = (__bf16*)alloc((size_t)H * D * DH * 2);
  __bf16* projT_f  = (__bf16*)alloc((size_t)H * D * DH * 2);
  __bf16* atteT    = (__bf16*)alloc((size_t)H * DH * DH * 2);
  __bf16* o_rgb    = (__bf16*)alloc(NTD * 2);    // (n,h,t,dh)
  __bf16* o_flow   = (__bf16*)alloc(NTD * 2);
  __bf16* oT_rgb   = (__bf16*)alloc(NTD * 2);    // (n,h,dh,t)
  __bf16* oT_flow  = (__bf16*)alloc(NTD * 2);
  __bf16* w_bf     = (__bf16*)alloc(NTD * 2);    // (n,h,t,dh)
  float*  att      = (float*) alloc((size_t)NH * T * T * 4);
  float*  rmax     = (float*) alloc((size_t)NH * T * 4);
  float*  rsum     = (float*) alloc((size_t)NH * T * 4);
  float*  cmax     = (float*) alloc((size_t)NH * T * 4);
  float*  csum     = (float*) alloc((size_t)NH * T * 4);
  __bf16* Pr       = (__bf16*)alloc((size_t)NH * T * T * 2);
  __bf16* Pf       = (__bf16*)alloc((size_t)NH * T * T * 2);
  float*  of32_a   = (float*) alloc(NTD * 4);    // O_rgb pre-norm, then e_rgb
  float*  of32_b   = (float*) alloc(NTD * 4);    // O_flow pre-norm, then e_flow

  const int EW = 256;
  // 1) converts / transposes
  cvt_xT_kernel<<<(int)(NTD / EW), EW, 0, stream>>>(rgb, xT_rgb);
  cvt_xT_kernel<<<(int)(NTD / EW), EW, 0, stream>>>(flow, xT_flow);
  cvt_wT_kernel<<<(H * D * DH) / EW, EW, 0, stream>>>(rgb_proj, projT_r, D, DH, H * D * DH);
  cvt_wT_kernel<<<(H * D * DH) / EW, EW, 0, stream>>>(flow_proj, projT_f, D, DH, H * D * DH);
  cvt_wT_kernel<<<(H * DH * DH) / EW, EW, 0, stream>>>(atte_w, atteT, DH, DH, H * DH * DH);

  // 2) projections: O = xT @ proj   (K = D)  [16x64 strips per wave]
  {
    int totalQ = NH * (T / 16) * (DH / 64);      // 16384 waves
    int blocks = totalQ / 8;
    gemm_kernel<0><<<blocks, 256, 0, stream>>>(xT_rgb, projT_r, of32_a,
        T / 16, DH / 64, D, D, D, DH,
        (long)T * D, H, (long)D * DH, H, (long)T * DH, totalQ);
    gemm_kernel<0><<<blocks, 256, 0, stream>>>(xT_flow, projT_f, of32_b,
        T / 16, DH / 64, D, D, D, DH,
        (long)T * D, H, (long)D * DH, H, (long)T * DH, totalQ);
  }
  // 3) L2 normalize -> bf16 O and O^T
  norm_kernel<<<NH * T, DH, 0, stream>>>(of32_a, o_rgb, oT_rgb);
  norm_kernel<<<NH * T, DH, 0, stream>>>(of32_b, o_flow, oT_flow);

  // 4) W = O_rgb @ atte_w  (K = DH), bf16 out
  {
    int totalQ = NH * (T / 16) * (DH / 64);
    gemm_kernel<1><<<totalQ / 8, 256, 0, stream>>>(o_rgb, atteT, w_bf,
        T / 16, DH / 64, DH, DH, DH, DH,
        (long)T * DH, 1, (long)DH * DH, H, (long)T * DH, totalQ);
  }
  // 5) att = W @ O_flow^T  (K = DH), TDM-staged LDS GEMM, f32 out
  {
    int blocks = NH * (T / 64) * (T / 128);      // 8192
    size_t shmem = (64 * DH + 128 * DH) * sizeof(__bf16);  // 96 KB
    att_gemm_kernel<<<blocks, 256, shmem, stream>>>(w_bf, o_flow, att);
  }
  // 6) softmax stats + probability matrices
  rowstats_kernel<<<NH * T, 256, 0, stream>>>(att, rmax, rsum);
  colstats_kernel<<<NH * T, 256, 0, stream>>>(att, cmax, csum);
  pmat_kernel<<<(int)(((long)NH * T * T) / EW), EW, 0, stream>>>(att, rmax, rsum,
                                                                 cmax, csum, Pr, Pf);
  // 7) e = gelu(P @ O)  (K = T), f32 out with exact-erf GELU epilogue
  {
    int totalQ = NH * (T / 16) * (DH / 64);
    gemm_kernel<2><<<totalQ / 8, 256, 0, stream>>>(Pr, oT_rgb, of32_a,
        T / 16, DH / 64, T, T, T, DH,
        (long)T * T, 1, (long)DH * T, NH, (long)T * DH, totalQ);
    gemm_kernel<2><<<totalQ / 8, 256, 0, stream>>>(Pf, oT_flow, of32_b,
        T / 16, DH / 64, T, T, T, DH,
        (long)T * T, 1, (long)DH * T, NH, (long)T * DH, totalQ);
  }
  // 8) out = tanh(e.mT.reshape + x), already in (n,d,t) layout
  final_kernel<<<(int)(NTD / EW), EW, 0, stream>>>(rgb, of32_a, out);
  final_kernel<<<(int)(NTD / EW), EW, 0, stream>>>(flow, of32_b, out + NTD);
}
